// AgentGnn_1202590843142
// MI455X (gfx1250) — matrix operand order repaired and checked by reading, not compiled, not run on hardware
//
#include <hip/hip_runtime.h>
#include <math.h>

#define N_NODES   32768
#define AGENTS    32
#define N_SAMPLES 1024
#define D         64
#define EPB       992   // edges per block = 32*31

typedef __attribute__((ext_vector_type(16))) _Float16 v16h;
typedef __attribute__((ext_vector_type(8)))  _Float16 v8h;
typedef __attribute__((ext_vector_type(8)))  float    v8f;
typedef __attribute__((ext_vector_type(4)))  float    v4f;
typedef int v4i __attribute__((vector_size(16)));     // matches builtin's param type

#if defined(__has_builtin)
#if __has_builtin(__builtin_amdgcn_global_load_async_to_lds_b128) && \
    __has_builtin(__builtin_amdgcn_s_wait_asynccnt)
#define USE_ASYNC_LDS 1
#endif
#endif
#ifndef USE_ASYNC_LDS
#define USE_ASYNC_LDS 0
#endif

__device__ __forceinline__ v16h cat16(v8h lo, v8h hi) {
    v16h r;
#pragma unroll
    for (int h = 0; h < 8; ++h) { r[h] = lo[h]; r[h + 8] = hi[h]; }
    return r;
}

// One workgroup per 32-agent sample block.
__global__ __launch_bounds__(128)
void cgconv_block_kernel(const float* __restrict__ x,
                         const float* __restrict__ edge_attr,
                         const float* __restrict__ Wf,
                         const float* __restrict__ bf,
                         const float* __restrict__ Ws,
                         const float* __restrict__ bs,
                         float* __restrict__ agg,
                         float* __restrict__ stats) {
    __shared__ __attribute__((aligned(16))) _Float16 sX[32][64];        //  4 KB
    __shared__ __attribute__((aligned(16))) _Float16 sWT[2][64][128];   // 32 KB (transposed, f16)
    __shared__ __attribute__((aligned(16))) float    sWe[2][2][64];     //  1 KB (edge rows 128,129)
    __shared__ __attribute__((aligned(16))) float    sB[4][64];         //  1 KB (wave-indexed bias, 1,3 = 0)
    __shared__ __attribute__((aligned(16))) _Float16 sP[4][32][64];     // 16 KB (Pfd,Pfs,Psd,Pss)
    __shared__ __attribute__((aligned(16))) float    sRed[2][64];       //  .5KB
    __shared__ __attribute__((aligned(16))) float    sEA[EPB * 2];      // 7.75 KB edge_attr block

    const int tid  = threadIdx.x;
    const int blk  = blockIdx.x;
    const int lane = tid & 31;
    const int wave = tid >> 5;
    const int hs   = lane >> 4;
    const int ln   = lane & 15;

    const float* ea = edge_attr + (size_t)blk * (EPB * 2);
    __builtin_prefetch(ea + tid * 15, 0, 0);   // global_prefetch_b8

    // ---- stage edge_attr block into LDS (async on gfx1250; overlaps GEMM) ----
#pragma unroll
    for (int it = 0; it < 4; ++it) {
        int chunk = tid + it * 128;           // 16-byte chunks, 496 total
        if (chunk < (EPB * 2) / 4) {
#if USE_ASYNC_LDS
            __builtin_amdgcn_global_load_async_to_lds_b128(
                (__attribute__((address_space(1))) v4i*)(ea + chunk * 4),
                (__attribute__((address_space(3))) v4i*)(sEA + chunk * 4),
                0, 0);
#else
            ((v4f*)sEA)[chunk] = ((const v4f*)ea)[chunk];
#endif
        }
    }

    // ---- stage X block as f16 ----
    const float* xblk = x + (size_t)blk * (AGENTS * D);
#pragma unroll
    for (int it = 0; it < 4; ++it) {
        int f4 = tid + it * 128;
        v4f v  = ((const v4f*)xblk)[f4];
        int r = (f4 * 4) >> 6, c = (f4 * 4) & 63;
        sX[r][c + 0] = (_Float16)v[0]; sX[r][c + 1] = (_Float16)v[1];
        sX[r][c + 2] = (_Float16)v[2]; sX[r][c + 3] = (_Float16)v[3];
    }
    // ---- stage weights (transposed, f16) + edge rows + biases ----
#pragma unroll
    for (int w = 0; w < 2; ++w) {
        const float* Wsrc = w ? Ws : Wf;
#pragma unroll 4
        for (int it = 0; it < 16; ++it) {
            int f4 = tid + it * 128;
            v4f v  = ((const v4f*)Wsrc)[f4];
            int k = (f4 * 4) >> 6, c = (f4 * 4) & 63;
            sWT[w][c + 0][k] = (_Float16)v[0];
            sWT[w][c + 1][k] = (_Float16)v[1];
            sWT[w][c + 2][k] = (_Float16)v[2];
            sWT[w][c + 3][k] = (_Float16)v[3];
        }
        { int r = tid >> 6, c = tid & 63; sWe[w][r][c] = Wsrc[(128 + r) * 64 + c]; }
    }
    if (tid < 64) {
        sB[0][tid] = bf[tid]; sB[1][tid] = 0.0f;
        sB[2][tid] = bs[tid]; sB[3][tid] = 0.0f;
    }
    { int r = tid >> 6, c = tid & 63; sRed[r][c] = 0.0f; }
    __syncthreads();

    // ---- WMMA: wave w computes one of {Pfd,Pfs,Psd,Pss} = X(32x64) @ W(64x64) ----
    {
        const int wSel  = wave >> 1;           // 0: Wf, 1: Ws
        const int kbase = (wave & 1) * 64;     // 0: dst rows, 64: src rows
        const v8f vzero = {0.f, 0.f, 0.f, 0.f, 0.f, 0.f, 0.f, 0.f};
#pragma unroll
        for (int mt = 0; mt < 2; ++mt) {
            v8f acc[4] = {vzero, vzero, vzero, vzero};
            const int m = mt * 16 + ln;
#pragma unroll
            for (int kt = 0; kt < 2; ++kt) {
                v8h alo = *(const v8h*)&sX[m][kt * 32 + hs * 16];
                v8h ahi = *(const v8h*)&sX[m][kt * 32 + hs * 16 + 8];
                v16h a  = cat16(alo, ahi);
#pragma unroll
                for (int nt = 0; nt < 4; ++nt) {
                    const int c = nt * 16 + ln;
                    const _Float16* bp = &sWT[wSel][c][kbase + kt * 32 + hs * 16];
                    v16h b = cat16(*(const v8h*)&bp[0], *(const v8h*)&bp[8]);
                    acc[nt] = __builtin_amdgcn_wmma_f32_16x16x32_f16(
                        false, a, false, b, (short)0, acc[nt], false, false);
                }
            }
#pragma unroll
            for (int nt = 0; nt < 4; ++nt) {
                const int   c  = nt * 16 + ln;
                const float bb = sB[wave][c];          // branch-free (rows 1,3 zero)
#pragma unroll
                for (int v = 0; v < 8; ++v) {          // C/D layout: M = v + 8*hs
                    int mr = mt * 16 + v + 8 * hs;
                    sP[wave][mr][c] = (_Float16)(acc[nt][v] + bb);
                }
            }
        }
    }
#if USE_ASYNC_LDS
    __builtin_amdgcn_s_wait_asynccnt(0);
#endif
    __syncthreads();

    // ---- edge gating + aggregation: thread = (dst node i, 16-channel group) ----
    const int di = tid >> 2;
    const int cg = tid & 3;
    const int c0 = cg * 16;

    float pfd[16], psd[16], wf0[16], wf1[16], ws0[16], ws1[16], accv[16];
#pragma unroll
    for (int cc = 0; cc < 16; ++cc) {
        pfd[cc] = (float)sP[0][di][c0 + cc];
        psd[cc] = (float)sP[2][di][c0 + cc];
        wf0[cc] = sWe[0][0][c0 + cc];
        wf1[cc] = sWe[0][1][c0 + cc];
        ws0[cc] = sWe[1][0][c0 + cc];
        ws1[cc] = sWe[1][1][c0 + cc];
        accv[cc] = 0.0f;
    }

    for (int j = 0; j < AGENTS; ++j) {
        if (j == di) continue;
        // meshgrid('ij') order: src-major, dst list skips dst==src
        const int    eidx = j * 31 + (di - (di > j ? 1 : 0));
        const float2 e    = ((const float2*)sEA)[eidx];
        _Float16 pfsh[16], pssh[16];
        *(v8h*)&pfsh[0] = *(const v8h*)&sP[1][j][c0];
        *(v8h*)&pfsh[8] = *(const v8h*)&sP[1][j][c0 + 8];
        *(v8h*)&pssh[0] = *(const v8h*)&sP[3][j][c0];
        *(v8h*)&pssh[8] = *(const v8h*)&sP[3][j][c0 + 8];
#pragma unroll
        for (int cc = 0; cc < 16; ++cc) {
            float f = pfd[cc] + (float)pfsh[cc] + e.x * wf0[cc] + e.y * wf1[cc];
            float s = psd[cc] + (float)pssh[cc] + e.x * ws0[cc] + e.y * ws1[cc];
            float sig = __builtin_amdgcn_rcpf(1.0f + __expf(-f));          // v_rcp_f32
            float sp  = fmaxf(s, 0.0f) + __logf(1.0f + __expf(-fabsf(s))); // v_log_f32
            accv[cc] += sig * sp;
        }
    }

    // write aggregated messages (pre-BN)
    float* aggp = agg + ((size_t)blk * AGENTS + di) * D + c0;
#pragma unroll
    for (int q = 0; q < 4; ++q) {
        v4f v = {accv[q * 4 + 0], accv[q * 4 + 1], accv[q * 4 + 2], accv[q * 4 + 3]};
        ((v4f*)aggp)[q] = v;
    }

    // BN partial stats: LDS f32 atomics, then one global atomic per channel
#pragma unroll
    for (int cc = 0; cc < 16; ++cc) {
        atomicAdd(&sRed[0][c0 + cc], accv[cc]);
        atomicAdd(&sRed[1][c0 + cc], accv[cc] * accv[cc]);
    }
    __syncthreads();
    if (tid < 64) {
        atomicAdd(&stats[tid],      sRed[0][tid]);
        atomicAdd(&stats[64 + tid], sRed[1][tid]);
    }
}

__global__ void bn_finalize_kernel(const float* __restrict__ stats,
                                   const float* __restrict__ gamma,
                                   const float* __restrict__ beta,
                                   float* __restrict__ sc) {
    int c = threadIdx.x;
    if (c < 64) {
        const float inv = 1.0f / (float)N_NODES;
        float mu  = stats[c] * inv;
        float var = stats[64 + c] * inv - mu * mu;
        float rs  = rsqrtf(var + 1e-5f);
        float a   = gamma[c] * rs;
        sc[c]      = a;
        sc[64 + c] = beta[c] - mu * a;
    }
}

__global__ __launch_bounds__(256)
void bn_apply_kernel(const float* __restrict__ agg,
                     const float* __restrict__ xin,
                     const float* __restrict__ sc,
                     float* __restrict__ out) {
    int idx = blockIdx.x * blockDim.x + threadIdx.x;  // float4 index
    int f   = idx * 4;
    if (f < N_NODES * D) {
        v4f a  = ((const v4f*)agg)[idx];
        v4f xv = ((const v4f*)xin)[idx];
        int c  = f & 63;
        v4f s  = *(const v4f*)&sc[c];
        v4f b  = *(const v4f*)&sc[64 + c];
        v4f r;
#pragma unroll
        for (int k = 0; k < 4; ++k) r[k] = fmaxf(a[k] * s[k] + b[k] + xv[k], 0.0f);
        ((v4f*)out)[idx] = r;
    }
}

extern "C" void kernel_launch(void* const* d_in, const int* in_sizes, int n_in,
                              void* d_out, int out_size, void* d_ws, size_t ws_size,
                              hipStream_t stream) {
    (void)in_sizes; (void)n_in; (void)out_size; (void)ws_size;
    const float* x         = (const float*)d_in[0];
    // d_in[1] = edge_index: structure is analytic (fully-connected 32-blocks), unused
    const float* edge_attr = (const float*)d_in[2];
    const float* Wf1 = (const float*)d_in[3];
    const float* bf1 = (const float*)d_in[4];
    const float* Ws1 = (const float*)d_in[5];
    const float* bs1 = (const float*)d_in[6];
    const float* g1  = (const float*)d_in[7];
    const float* be1 = (const float*)d_in[8];
    const float* Wf2 = (const float*)d_in[9];
    const float* bf2 = (const float*)d_in[10];
    const float* Ws2 = (const float*)d_in[11];
    const float* bs2 = (const float*)d_in[12];
    const float* g2  = (const float*)d_in[13];
    const float* be2 = (const float*)d_in[14];

    float* ws  = (float*)d_ws;
    float* agg = ws;                              // N_NODES*D floats
    float* h   = ws + (size_t)N_NODES * D;        // N_NODES*D floats
    float* st1 = ws + (size_t)2 * N_NODES * D;    // 128 (sum, sumsq)
    float* sc1 = st1 + 128;                       // 128 (scale, shift)
    float* st2 = sc1 + 128;
    float* sc2 = st2 + 128;

    (void)hipMemsetAsync(st1, 0, 512 * sizeof(float), stream);

    const int applyBlocks = (N_NODES * D / 4 + 255) / 256;

    // layer 1
    cgconv_block_kernel<<<N_SAMPLES, 128, 0, stream>>>(x, edge_attr, Wf1, bf1, Ws1, bs1, agg, st1);
    bn_finalize_kernel<<<1, 64, 0, stream>>>(st1, g1, be1, sc1);
    bn_apply_kernel<<<applyBlocks, 256, 0, stream>>>(agg, x, sc1, h);
    // layer 2
    cgconv_block_kernel<<<N_SAMPLES, 128, 0, stream>>>(h, edge_attr, Wf2, bf2, Ws2, bs2, agg, st2);
    bn_finalize_kernel<<<1, 64, 0, stream>>>(st2, g2, be2, sc2);
    bn_apply_kernel<<<applyBlocks, 256, 0, stream>>>(agg, h, sc2, (float*)d_out);
}
